// SeparableAttn_2310692405621
// MI455X (gfx1250) — matrix operand order repaired and checked by reading, not compiled, not run on hardware
//
#include <hip/hip_runtime.h>
#include <math.h>

typedef _Float16 f16;
typedef __attribute__((ext_vector_type(16))) _Float16 v16h;
typedef __attribute__((ext_vector_type(8)))  _Float16 v8h;
typedef __attribute__((ext_vector_type(8)))  float    v8f;
typedef __attribute__((ext_vector_type(4)))  float    v4f;

#define BB 2
#define CC 256
#define C2c 128
#define SS 65536            // T*W*H = 16*64*64; channel stride = 1<<16
#define LSS 16

// ---------------------------------------------------------------------------
// Pack fp32 weights/biases into concatenated f16 matrix [3][512][256]:
// rows 0..127 = Wq, 128..255 = Wk, 256..511 = Wv per cell; bias alongside.
// ---------------------------------------------------------------------------
__global__ void pack_weights_kernel(const float* __restrict__ Wq, const float* __restrict__ bq,
                                    const float* __restrict__ Wk, const float* __restrict__ bk,
                                    const float* __restrict__ Wv, const float* __restrict__ bv,
                                    f16* __restrict__ Wall, float* __restrict__ Ball)
{
    int idx  = blockIdx.x * 256 + threadIdx.x;   // 3*512*256 total
    int k    = idx & 255;
    int r    = (idx >> 8) & 511;
    int cell = idx >> 17;
    float w;
    if (r < 128)      w = Wq[(cell * 128 + r) * 256 + k];
    else if (r < 256) w = Wk[(cell * 128 + (r - 128)) * 256 + k];
    else              w = Wv[(cell * 256 + (r - 256)) * 256 + k];
    Wall[idx] = (f16)w;
    if (k == 0) {
        float bb;
        if (r < 128)      bb = bq[cell * 128 + r];
        else if (r < 256) bb = bk[cell * 128 + (r - 128)];
        else              bb = bv[cell * 256 + (r - 256)];
        Ball[cell * 512 + r] = bb;
    }
}

// ---------------------------------------------------------------------------
// Fused QKV 1x1-conv GEMM: Y[512 x 64] = Wall[512 x 256] * X[256 x 64] + bias
// per (batch, 64-position tile).  WMMA f32_16x16x32_f16, dual accumulator
// chains, branchless permuted scatter into Y[B][512][S] (f16).
//   a = (p>>sa)&(A-1), i = (p>>si)&(d1-1), jj = (p>>sj)&(d2-1)
// ---------------------------------------------------------------------------
__global__ __launch_bounds__(256)
void conv_qkv_kernel(const float* __restrict__ xsrc,
                     const f16*  __restrict__ Wc,    // [512][256] f16
                     const float* __restrict__ Bc,   // [512] bias
                     f16* __restrict__ Y,            // [B][512][A][d1][d2]
                     int sa, int si, int sj,
                     int lA, int ld1, int ld2)
{
    __shared__ f16 xs[64][264];                     // X tile transposed [p][c], padded
    const int tid = threadIdx.x;
    const int blocksPerBatch = SS / 64;
    const int b  = blockIdx.x / blocksPerBatch;
    const int p0 = (blockIdx.x % blocksPerBatch) * 64;

    // stage fp32 X tile -> f16 LDS, float4 global loads (fully coalesced)
    {
        const float* xb = xsrc + ((size_t)b * CC << LSS) + p0;
        const int pl = (tid & 15) * 4;
        for (int e = 0; e < 16; ++e) {
            const int c = e * 16 + (tid >> 4);      // 16 channels / iter
            const v4f v = *(const v4f*)(xb + ((size_t)c << LSS) + pl);
            xs[pl + 0][c] = (f16)v.x;
            xs[pl + 1][c] = (f16)v.y;
            xs[pl + 2][c] = (f16)v.z;
            xs[pl + 3][c] = (f16)v.w;
        }
    }
    __syncthreads();

    const int wave = tid >> 5;
    const int lane = tid & 31;
    const int ln   = lane & 15;
    const int hi   = lane >> 4;

    for (int mm = 0; mm < 4; ++mm) {
        const int cobase = (wave * 4 + mm) * 16;    // 8 waves x 4 = 32 M-tiles

        // A-fragments (weights) for all 8 K-steps, kept in registers
        v16h af[8];
        {
            const f16* wrow = Wc + (size_t)(cobase + ln) * 256;
            #pragma unroll
            for (int ks = 0; ks < 8; ++ks) {
                const int k0 = ks * 32;
                v8h lo  = *(const v8h*)(wrow + k0 + hi * 8);
                v8h hi8 = *(const v8h*)(wrow + k0 + 16 + hi * 8);
                af[ks]  = __builtin_shufflevector(lo, hi8,
                              0,1,2,3,4,5,6,7,8,9,10,11,12,13,14,15);
            }
        }
        const v8f bias = *(const v8f*)(Bc + cobase + hi * 8);

        #pragma unroll
        for (int ntp = 0; ntp < 2; ++ntp) {         // ntile pairs: 2 indep chains
            const int nt0 = ntp * 2, nt1 = nt0 + 1;
            v8f acc0 = bias, acc1 = bias;
            #pragma unroll
            for (int ks = 0; ks < 8; ++ks) {
                const int k0 = ks * 32 + hi * 8;
                const f16* r0 = &xs[nt0 * 16 + ln][k0];
                const f16* r1 = &xs[nt1 * 16 + ln][k0];
                v8h l0 = *(const v8h*)(r0), h0 = *(const v8h*)(r0 + 16);
                v8h l1 = *(const v8h*)(r1), h1 = *(const v8h*)(r1 + 16);
                v16h b0 = __builtin_shufflevector(l0, h0,
                              0,1,2,3,4,5,6,7,8,9,10,11,12,13,14,15);
                v16h b1 = __builtin_shufflevector(l1, h1,
                              0,1,2,3,4,5,6,7,8,9,10,11,12,13,14,15);
                acc0 = __builtin_amdgcn_wmma_f32_16x16x32_f16(
                           false, af[ks], false, b0, (short)0, acc0, false, false);
                acc1 = __builtin_amdgcn_wmma_f32_16x16x32_f16(
                           false, af[ks], false, b1, (short)0, acc1, false, false);
            }
            // branchless permuted scatter (channel stride = S = 1<<16)
            #pragma unroll
            for (int half = 0; half < 2; ++half) {
                const v8f acc = half ? acc1 : acc0;
                const int p = p0 + (half ? nt1 : nt0) * 16 + ln;
                const int a  = (p >> sa) & ((1 << lA) - 1);
                const int i  = (p >> si) & ((1 << ld1) - 1);
                const int jj = (p >> sj) & ((1 << ld2) - 1);
                const size_t off_sp = ((size_t)(((a << ld1) + i) << ld2)) + (size_t)jj;
                f16* yb = Y + (((size_t)(b * 512 + cobase + hi * 8)) << LSS) + off_sp;
                #pragma unroll
                for (int r = 0; r < 8; ++r)
                    yb[(size_t)r << LSS] = (f16)acc[r];
            }
        }
    }
}

// ---------------------------------------------------------------------------
// MaxPool(2,1,1) along attention axis for V (Y rows 256..511):
// vp flat = v_r[x'][p'] row-major.
// ---------------------------------------------------------------------------
__global__ void pool_v_kernel(const f16* __restrict__ Y, f16* __restrict__ vp, int ld12)
{
    const int o  = blockIdx.x * 256 + threadIdx.x;  // B*C*S/2 = 1<<24
    const int b  = o >> 23;
    const int ob = o & ((1 << 23) - 1);
    const int c  = ob >> 15;                        // pooled ch block = S/2
    const int r  = ob & 32767;
    const int pa = r >> ld12;
    const int ij = r & ((1 << ld12) - 1);
    const size_t in = (((size_t)(b * 512 + 256 + c)) << LSS)
                    + ((size_t)(2 * pa) << ld12) + (size_t)ij;
    const float f1 = (float)Y[in];
    const float f2 = (float)Y[in + (1 << ld12)];
    vp[o] = (f16)fmaxf(f1, f2);
}

// ---------------------------------------------------------------------------
// MaxPool + transpose for K (Y rows 128..255): kt[b][p'][j'] = k_r[j'][p'].
// ---------------------------------------------------------------------------
__global__ void pool_tk_kernel(const f16* __restrict__ Y, f16* __restrict__ kt,
                               int lJ, int lA2, int ld12)
{
    const int o  = blockIdx.x * 256 + threadIdx.x;  // B*C2*S/2 = 1<<23
    const int b  = o >> 22;
    const int ob = o & ((1 << 22) - 1);
    const int pp = ob >> lJ;                        // p'
    const int j  = ob & ((1 << lJ) - 1);            // j'
    const int g  = (j << lA2) + pp;                 // pooled [C2][A/2][d1][d2] flat
    const int c2 = g >> 15;
    const int r  = g & 32767;
    const int pa = r >> ld12;
    const int ij = r & ((1 << ld12) - 1);
    const size_t in = (((size_t)(b * 512 + 128 + c2)) << LSS)
                    + ((size_t)(2 * pa) << ld12) + (size_t)ij;
    const float f1 = (float)Y[in];
    const float f2 = (float)Y[in + (1 << ld12)];
    kt[o] = (f16)fmaxf(f1, f2);
}

__global__ void init_logits_kernel(float* __restrict__ logits)
{
    logits[blockIdx.x * 256 + threadIdx.x] = 0.0f;
}

// ---------------------------------------------------------------------------
// logits[A x A/2] += q_r[A x J] * k_r[J x A/2]; split-K WMMA (dual chains)
// + f32 global atomics.  One wave per (b, m-tile, n-tile, K-chunk of 4096).
// Lanes with n >= A/2 load a clamped (in-bounds) k row; their D columns are
// garbage but discarded at the guarded atomicAdd, so EXEC stays full.
// ---------------------------------------------------------------------------
__global__ __launch_bounds__(32)
void logits_kernel(const f16* __restrict__ Y, const f16* __restrict__ kt,
                   float* __restrict__ logits, int Mt, int Nt, int KC, int lJ, int A2)
{
    const int lane = threadIdx.x;
    const int ln = lane & 15, hi = lane >> 4;
    int bx = blockIdx.x;
    const int kc = bx % KC; bx /= KC;
    const int n2 = bx % Nt; bx /= Nt;
    const int m  = bx % Mt; bx /= Mt;
    const int b  = bx;
    const int J  = 1 << lJ;

    // q rows live in Y rows 0..127 (contiguous A*J per batch)
    const f16* qrow = Y + (((size_t)b * 512) << LSS) + (size_t)(m * 16 + ln) * J;
    const int  n      = n2 * 16 + ln;
    const bool nvalid = (n < A2);
    const f16* krow = kt + ((size_t)b * (C2c * (SS / 2)))
                         + (size_t)(nvalid ? n : (A2 - 1)) * J;

    v8f acc0 = {}, acc1 = {};
    const int KCH = J / KC;                 // 4096
    const int j0b = kc * KCH;
    for (int j0 = j0b; j0 < j0b + KCH; j0 += 64) {
        v8h a0l = *(const v8h*)(qrow + j0 + hi * 8);
        v8h a0h = *(const v8h*)(qrow + j0 + 16 + hi * 8);
        v8h a1l = *(const v8h*)(qrow + j0 + 32 + hi * 8);
        v8h a1h = *(const v8h*)(qrow + j0 + 48 + hi * 8);
        v8h b0l = *(const v8h*)(krow + j0 + hi * 8);
        v8h b0h = *(const v8h*)(krow + j0 + 16 + hi * 8);
        v8h b1l = *(const v8h*)(krow + j0 + 32 + hi * 8);
        v8h b1h = *(const v8h*)(krow + j0 + 48 + hi * 8);
        v16h a0 = __builtin_shufflevector(a0l, a0h, 0,1,2,3,4,5,6,7,8,9,10,11,12,13,14,15);
        v16h a1 = __builtin_shufflevector(a1l, a1h, 0,1,2,3,4,5,6,7,8,9,10,11,12,13,14,15);
        v16h b0 = __builtin_shufflevector(b0l, b0h, 0,1,2,3,4,5,6,7,8,9,10,11,12,13,14,15);
        v16h b1 = __builtin_shufflevector(b1l, b1h, 0,1,2,3,4,5,6,7,8,9,10,11,12,13,14,15);
        acc0 = __builtin_amdgcn_wmma_f32_16x16x32_f16(
                   false, a0, false, b0, (short)0, acc0, false, false);
        acc1 = __builtin_amdgcn_wmma_f32_16x16x32_f16(
                   false, a1, false, b1, (short)0, acc1, false, false);
    }
    const v8f acc = acc0 + acc1;
    if (nvalid) {
        #pragma unroll
        for (int r = 0; r < 8; ++r) {
            const int row = m * 16 + hi * 8 + r;
            atomicAdd(&logits[(size_t)(b * 64 + row) * 32 + n], acc[r]);
        }
    }
}

// ---------------------------------------------------------------------------
// Row softmax over A/2 logits (padded pitch 32); one wave per (b,a).
// ---------------------------------------------------------------------------
__global__ __launch_bounds__(32)
void softmax_kernel(const float* __restrict__ logits, float* __restrict__ attn,
                    int A, int A2)
{
    const int row  = blockIdx.x;            // b*A + a
    const int b    = row / A;
    const int a    = row - b * A;
    const int lane = threadIdx.x;
    const size_t base = (size_t)(b * 64 + a) * 32;
    float v = (lane < A2) ? logits[base + lane] : -1e30f;
    float mx = v;
    for (int s = 16; s > 0; s >>= 1) mx = fmaxf(mx, __shfl_xor(mx, s, 32));
    float e = (lane < A2) ? __expf(v - mx) : 0.0f;
    float sum = e;
    for (int s = 16; s > 0; s >>= 1) sum += __shfl_xor(sum, s, 32);
    attn[base + lane] = (lane < A2) ? (e / sum) : 0.0f;
}

// ---------------------------------------------------------------------------
// Epilogue: o[x',a] = sum_p v_r[x',p] * attn[a,p]; dst = gamma*o + resid.
// Branchless inverse permutation: pout = (i<<s_i) + (jj<<s_j) + (a<<s_a)
// (covers the H-cell's scrambled raw view with s_i=10, s_j=6, s_a=0).
// ---------------------------------------------------------------------------
__global__ __launch_bounds__(256)
void output_kernel(const float* __restrict__ resid,
                   const f16*  __restrict__ vp,
                   const float* __restrict__ attn,
                   const float* __restrict__ gamma_ptr,
                   float* __restrict__ dst,
                   int cell, int lA, int lA2, int ld12, int ld2, int lX,
                   int s_a, int s_i, int s_j)
{
    __shared__ float at_s[64 * 32];
    const int A  = 1 << lA;
    const int A2 = 1 << lA2;
    const int tid = threadIdx.x;
    const int bpb = 1 << (lX - 8);
    const int b   = blockIdx.x / bpb;
    const int x   = ((blockIdx.x - b * bpb) << 8) + tid;
    const float g = gamma_ptr[cell];

    for (int idx = tid; idx < A * 32; idx += 256)
        at_s[idx] = attn[(size_t)b * 64 * 32 + idx];
    __syncthreads();

    float vr[32];
    {
        const f16* vrow = vp + (size_t)b * (CC * (SS / 2)) + ((size_t)x << lA2);
        for (int p8 = 0; p8 < (A2 >> 3); ++p8) {
            const v8h v = *(const v8h*)(vrow + p8 * 8);
            #pragma unroll
            for (int q = 0; q < 8; ++q) vr[p8 * 8 + q] = (float)v[q];
        }
    }
    const int rr = x & ((1 << ld12) - 1);   // i*d2 + jj
    const int i  = rr >> ld2;
    const int jj = rr & ((1 << ld2) - 1);
    const int pbase = (i << s_i) + (jj << s_j);
    const size_t chbase = ((size_t)b * CC + (x >> ld12)) << LSS;

    for (int a = 0; a < A; ++a) {
        float o = 0.0f;
        for (int p = 0; p < A2; ++p) o += vr[p] * at_s[(a << 5) + p];
        const size_t idx = chbase + (size_t)(pbase + (a << s_a));
        dst[idx] = g * o + resid[idx];
    }
}

// ---------------------------------------------------------------------------
extern "C" void kernel_launch(void* const* d_in, const int* in_sizes, int n_in,
                              void* d_out, int out_size, void* d_ws, size_t ws_size,
                              hipStream_t stream)
{
    (void)in_sizes; (void)n_in; (void)out_size; (void)ws_size;
    const float* x     = (const float*)d_in[0];
    const float* Wq    = (const float*)d_in[1];
    const float* bq    = (const float*)d_in[2];
    const float* Wk    = (const float*)d_in[3];
    const float* bk    = (const float*)d_in[4];
    const float* Wv    = (const float*)d_in[5];
    const float* bv    = (const float*)d_in[6];
    const float* gamma = (const float*)d_in[7];
    float* out = (float*)d_out;

    char* ws = (char*)d_ws;
    size_t off = 0;
    auto alloc = [&](size_t bytes) -> void* {
        void* p = ws + off;
        off = (off + bytes + 255) & ~(size_t)255;
        return p;
    };
    float* xbuf   = (float*)alloc((size_t)BB * CC * SS * 4);        // ping buffer
    f16*   Wall   = (f16*)  alloc((size_t)3 * 512 * 256 * 2);
    float* Ball   = (float*)alloc((size_t)3 * 512 * 4);
    f16*   Y      = (f16*)  alloc((size_t)BB * 512 * SS * 2);       // fused q|k|v
    f16*   kt     = (f16*)  alloc((size_t)BB * C2c * (SS / 2) * 2); // k pooled+transposed
    f16*   vp     = (f16*)  alloc((size_t)BB * CC  * (SS / 2) * 2); // v pooled
    float* logits = (float*)alloc((size_t)BB * 64 * 32 * 4);
    float* attn   = (float*)alloc((size_t)BB * 64 * 32 * 4);

    pack_weights_kernel<<<(3 * 512 * 256) / 256, 256, 0, stream>>>(
        Wq, bq, Wk, bk, Wv, bv, Wall, Ball);

    //           lA ld1 ld2 lJ  sa si sj  s_a s_i s_j
    struct Cfg { int lA, ld1, ld2, lJ, sa, si, sj, s_a, s_i, s_j; };
    const Cfg cfg[3] = {
        {4, 6, 6, 19, 12,  6, 0, 12,  6, 0},   // T: a=t, i=w, jj=h
        {6, 4, 6, 17,  6, 12, 0,  6, 12, 0},   // W: a=w, i=t, jj=h
        {6, 6, 4, 17,  0,  6, 12, 0, 10, 6},   // H: a=h, i=w, jj=t (raw-view scramble)
    };

    for (int cell = 0; cell < 3; ++cell) {
        const Cfg cg = cfg[cell];
        const int A    = 1 << cg.lA;
        const int A2   = A >> 1;
        const int lA2  = cg.lA - 1;
        const int ld12 = cg.ld1 + cg.ld2;
        const int lX   = 8 + ld12;          // X = C * d1 * d2
        const int J    = 1 << cg.lJ;

        const float* src = (cell == 0) ? x : xbuf;
        float* dstbuf = (cell == 2) ? out : xbuf;

        conv_qkv_kernel<<<BB * (SS / 64), 256, 0, stream>>>(
            src, Wall + (size_t)cell * 512 * 256, Ball + cell * 512,
            Y, cg.sa, cg.si, cg.sj, cg.lA, cg.ld1, cg.ld2);

        pool_v_kernel<<<(BB * CC * (SS / 2)) / 256, 256, 0, stream>>>(Y, vp, ld12);
        pool_tk_kernel<<<(BB * C2c * (SS / 2)) / 256, 256, 0, stream>>>(
            Y, kt, cg.lJ, lA2, ld12);

        init_logits_kernel<<<(BB * 64 * 32) / 256, 256, 0, stream>>>(logits);
        const int Mt = A / 16;
        const int Nt = (A2 + 15) / 16;
        const int KC = J / 4096;
        logits_kernel<<<BB * Mt * Nt * KC, 32, 0, stream>>>(
            Y, kt, logits, Mt, Nt, KC, cg.lJ, A2);

        softmax_kernel<<<BB * A, 32, 0, stream>>>(logits, attn, A, A2);

        output_kernel<<<BB * (1 << (lX - 8)), 256, 0, stream>>>(
            src, vp, attn, gamma, dstbuf, cell, cg.lA, lA2, ld12, cg.ld2, lX,
            cg.s_a, cg.s_i, cg.s_j);
    }
}
